// TimeMixing_13941463842954
// MI455X (gfx1250) — matrix operand order repaired and checked by reading, not compiled, not run on hardware
//
#include <hip/hip_runtime.h>

// RWKV time-mix for MI455X (gfx1250, wave32).
//   B = 16384 tokens, S = 1024 hidden.
// Prep   : Wk/Wv/Wr/Wout f32 -> bf16 once (workspace), so GEMM staging can use
//          ASYNC global->LDS copies (no VALU, overlaps WMMA).
// Kernel1: k/v/r GEMMs fused (bf16 WMMA, f32 accum) + full nonlinearity,
//          writes z = sigmoid(r)*wkv as bf16 into workspace.
//          A-tiles: token-shift mix in VALU (co-executes with WMMA);
//          W-tiles: async copies, double-buffered LDS.
// Kernel2: y = z @ Wout^T + bout; staging 100% async, double-buffered.

#define S_DIM 1024
#define BM    128        // token rows per block
#define BK    32         // K slice per stage (= WMMA K)
#define NT    (S_DIM / BK)
#define LDT   (BK + 8)   // padded LDS row stride: 40 elts = 80 B = 5*16 B

typedef __attribute__((ext_vector_type(16))) __bf16 v16b;
typedef __attribute__((ext_vector_type(8)))  float  v8f;
typedef int v4i_vs __attribute__((vector_size(16)));   // matches builtin param

union Frag { v16b v; uint4 q[2]; };

#define GAS __attribute__((address_space(1)))
#define LAS __attribute__((address_space(3)))

// ---- CDNA5 async global->LDS (ASYNCcnt path), guarded w/ safe fallback ----
__device__ __forceinline__ void async_b128(const unsigned short* g,
                                           unsigned short* l) {
#if __has_builtin(__builtin_amdgcn_global_load_async_to_lds_b128)
  __builtin_amdgcn_global_load_async_to_lds_b128(
      (GAS v4i_vs*)g, (LAS v4i_vs*)l, 0, 0);
#else
  *(uint4*)l = *(const uint4*)g;
#endif
}

__device__ __forceinline__ void wait_async0() {
#if __has_builtin(__builtin_amdgcn_s_wait_asynccnt)
  __builtin_amdgcn_s_wait_asynccnt(0);
#else
  asm volatile("s_wait_asynccnt 0x0" ::: "memory");
#endif
}

// round-to-nearest-even f32 -> bf16, packed pair into one dword
__device__ __forceinline__ unsigned pk_bf16(float lo, float hi) {
  unsigned a = __float_as_uint(lo);
  unsigned b = __float_as_uint(hi);
  a = (a + 0x7FFFu + ((a >> 16) & 1u)) >> 16;
  b = (b + 0x7FFFu + ((b >> 16) & 1u)) & 0xFFFF0000u;
  return b | a;
}

__device__ __forceinline__ unsigned short f2bf(float f) {
  unsigned u = __float_as_uint(f);
  return (unsigned short)((u + 0x7FFFu + ((u >> 16) & 1u)) >> 16);
}

__device__ __forceinline__ v8f wmma_bf16(v16b a, v16b b, v8f c) {
  return __builtin_amdgcn_wmma_f32_16x16x32_bf16(false, a, false, b, (short)0, c,
                                                 false, false);
}

// token-shift mix of 4 lanes + bf16 pack:  x*m + lx*(1-m) = lx + (x-lx)*m
__device__ __forceinline__ uint2 mix_pack(float4 xv, float4 m, float4 lx) {
  float e0 = fmaf(xv.x - lx.x, m.x, lx.x);
  float e1 = fmaf(xv.y - lx.y, m.y, lx.y);
  float e2 = fmaf(xv.z - lx.z, m.z, lx.z);
  float e3 = fmaf(xv.w - lx.w, m.w, lx.w);
  uint2 p;
  p.x = pk_bf16(e0, e1);
  p.y = pk_bf16(e2, e3);
  return p;
}

// ---------------------------------------------------------------------------
// Prep: f32 weight matrix -> bf16 (run once per weight; 1024 blocks x 256)
// ---------------------------------------------------------------------------
__global__ __launch_bounds__(256, 4)
void cvt_bf16_kernel(const float* __restrict__ src,
                     unsigned short* __restrict__ dst) {
  const size_t i = ((size_t)blockIdx.x * 256 + threadIdx.x) * 4;
  const float4 v = *(const float4*)(src + i);
  uint2 p;
  p.x = pk_bf16(v.x, v.y);
  p.y = pk_bf16(v.z, v.w);
  *(uint2*)(dst + i) = p;
}

// ---------------------------------------------------------------------------
// Kernel 1: fused k/v/r GEMMs + RWKV nonlinearity -> z (bf16)
// 256 thr = 8 waves (4 M-waves x 2 N-waves), block tile 128x32,
// double-buffered LDS; weights staged via async copies.
// ---------------------------------------------------------------------------
__global__ __launch_bounds__(256, 2)
void rwkv_kvr_kernel(const float* __restrict__ x,
                     const float* __restrict__ last_x,
                     const float* __restrict__ last_num,
                     const float* __restrict__ last_den,
                     const float* __restrict__ bonus,
                     const float* __restrict__ mix_k,
                     const float* __restrict__ mix_v,
                     const float* __restrict__ mix_r,
                     const unsigned short* __restrict__ wkb,
                     const unsigned short* __restrict__ wvb,
                     const unsigned short* __restrict__ wrb,
                     const float* __restrict__ bk,
                     const float* __restrict__ bv,
                     const float* __restrict__ br,
                     unsigned short* __restrict__ z)
{
  __shared__ unsigned short lx[2][3][BM * LDT];   // mixed activations (k,v,r)
  __shared__ unsigned short lw[2][3][32 * LDT];   // weight tiles  (k,v,r)

  const int t       = threadIdx.x;
  const int rowBase = blockIdx.y * BM;
  const int jBase   = blockIdx.x * 32;

  // staging decomposition (A path): 8 col-groups of 4 floats, 32 rows/pass
  const int c4 = (t & 7) * 4;
  const int r0 = t >> 3;
  // staging decomposition (W async path): 32 rows x 4 b128 chunks per weight
  const int wr_ = (t >> 2) & 31;
  const int wc_ = (t & 3) * 8;

  // wave / lane fragment decomposition (ISA 7.12.2 layouts)
  const int lane = t & 31;
  const int wid  = t >> 5;
  const int wM   = wid & 3;            // 4 M-waves * 32 rows
  const int wN   = wid >> 2;           // 2 N-waves * 16 cols
  const int am   = lane & 15;
  const int ak   = (lane >> 4) * 8;    // A: lanes 0-15 K{0-7,16-23}; 16-31 K{8-15,24-31}
  const int bn   = lane & 15;
  const int bk16 = (lane >> 4) * 8;    // B: lanes 0-15 K{0-7,16-23}; 16-31 K{8-15,24-31}

  v8f cK[2] = {};
  v8f cV[2] = {};
  v8f cR[2] = {};

  // ---- stage one K-tile into buffer `buf` ----
  auto stage = [&](int kt, int buf) {
    const int k0 = kt * BK;
    // A tiles: token-shift mixing, f32 -> bf16, ds_store (VALU path)
    const float4 mk = *(const float4*)(mix_k  + k0 + c4);
    const float4 mv = *(const float4*)(mix_v  + k0 + c4);
    const float4 mr = *(const float4*)(mix_r  + k0 + c4);
    const float4 lxv4 = *(const float4*)(last_x + k0 + c4);
#pragma unroll
    for (int it = 0; it < 4; ++it) {
      const int    row = r0 + it * 32;
      const size_t go  = (size_t)(rowBase + row) * S_DIM + k0 + c4;
      const float4 xv4 = *(const float4*)(x + go);
      __builtin_prefetch(x + go + BK, 0, 3);   // next K slice
      *(uint2*)&lx[buf][0][row * LDT + c4] = mix_pack(xv4, mk, lxv4);
      *(uint2*)&lx[buf][1][row * LDT + c4] = mix_pack(xv4, mv, lxv4);
      *(uint2*)&lx[buf][2][row * LDT + c4] = mix_pack(xv4, mr, lxv4);
    }
    // W tiles (already bf16): async global->LDS, no VALU / no VGPR data
    const size_t wgo = (size_t)(jBase + wr_) * S_DIM + k0 + wc_;
    const int    wlo = wr_ * LDT + wc_;
    const int    w01 = (t >> 7) & 1;                 // t<128 -> Wk, else Wv
    async_b128(((t & 128) ? wvb : wkb) + wgo, &lw[buf][w01][wlo]);
    if (t < 128)
      async_b128(wrb + wgo, &lw[buf][2][wlo]);
  };

  // ---- compute one K-tile from buffer `buf`: 6 WMMAs per wave ----
  auto compute = [&](int buf) {
    Frag bK, bV, bR;
    {
      const int wb = (wN * 16 + bn) * LDT + bk16;
      bK.q[0] = *(const uint4*)&lw[buf][0][wb];
      bK.q[1] = *(const uint4*)&lw[buf][0][wb + 16];
      bV.q[0] = *(const uint4*)&lw[buf][1][wb];
      bV.q[1] = *(const uint4*)&lw[buf][1][wb + 16];
      bR.q[0] = *(const uint4*)&lw[buf][2][wb];
      bR.q[1] = *(const uint4*)&lw[buf][2][wb + 16];
    }
#pragma unroll
    for (int mi = 0; mi < 2; ++mi) {
      const int arow = (wM * 32 + mi * 16 + am) * LDT + ak;
      Frag aK, aV, aR;
      aK.q[0] = *(const uint4*)&lx[buf][0][arow];
      aK.q[1] = *(const uint4*)&lx[buf][0][arow + 16];
      aV.q[0] = *(const uint4*)&lx[buf][1][arow];
      aV.q[1] = *(const uint4*)&lx[buf][1][arow + 16];
      aR.q[0] = *(const uint4*)&lx[buf][2][arow];
      aR.q[1] = *(const uint4*)&lx[buf][2][arow + 16];
      cK[mi] = wmma_bf16(aK.v, bK.v, cK[mi]);
      cV[mi] = wmma_bf16(aV.v, bV.v, cV[mi]);
      cR[mi] = wmma_bf16(aR.v, bR.v, cR[mi]);
    }
  };

  // ---- software pipeline: stage(kt+1) overlaps compute(kt) ----
  stage(0, 0);
  for (int kt = 0; kt < NT; ++kt) {
    const int buf = kt & 1;
    wait_async0();        // drain this wave's async copies for `buf`
    __syncthreads();      // + cross-wave ds_store visibility
    if (kt + 1 < NT) stage(kt + 1, buf ^ 1);
    compute(buf);
  }

  // ---- epilogue: bias + exp/wkv/sigmoid, write z as bf16 ----
  const int   j   = jBase + wN * 16 + bn;
  const float bkj = bk[j],    bvj = bv[j],       brj = br[j];
  const float boj = bonus[j], lnj = last_num[j], ldj = last_den[j];
  const int   mh  = (lane >> 4) * 8;
#pragma unroll
  for (int mi = 0; mi < 2; ++mi) {
#pragma unroll
    for (int vi = 0; vi < 8; ++vi) {
      const int   row = rowBase + wM * 32 + mi * 16 + mh + vi;
      const float kk  = cK[mi][vi] + bkj;
      const float vv  = cV[mi][vi] + bvj;
      const float rr  = cR[mi][vi] + brj;
      const float e   = __expf(boj + kk);
      const float wkv = fmaf(e, vv, lnj) / (ldj + e);
      const float sig = 1.0f / (1.0f + __expf(-rr));
      z[(size_t)row * S_DIM + j] = f2bf(sig * wkv);
    }
  }
}

// ---------------------------------------------------------------------------
// Kernel 2: y = z @ Wout^T + bout.  tile 128x64, waves 4Mx2N, wave tile 32x32.
// All staging is async (z and Wout already bf16), double-buffered LDS.
// ---------------------------------------------------------------------------
__global__ __launch_bounds__(256, 2)
void rwkv_out_kernel(const unsigned short* __restrict__ z,
                     const unsigned short* __restrict__ wob,
                     const float* __restrict__ bout,
                     float* __restrict__ y)
{
  __shared__ unsigned short lza[2][BM * LDT];
  __shared__ unsigned short lwo[2][64 * LDT];

  const int t       = threadIdx.x;
  const int rowBase = blockIdx.y * BM;
  const int jBase   = blockIdx.x * 64;

  const int sr = t >> 2;         // 0..63 staging row
  const int sc = (t & 3) * 8;    // b128 chunk within a 32-elt row

  const int lane = t & 31;
  const int wid  = t >> 5;
  const int wM   = wid & 3;
  const int wN   = wid >> 2;
  const int am   = lane & 15;
  const int ak   = (lane >> 4) * 8;
  const int bn   = lane & 15;
  const int bk16 = (lane >> 4) * 8;

  v8f acc[2][2] = {};

  auto stage = [&](int kt, int buf) {
    const int k0 = kt * BK;
    // A (z, bf16): 128 rows x 32 cols -> 512 b128 chunks, 2 per thread
    async_b128(z + (size_t)(rowBase + sr) * S_DIM + k0 + sc,
               &lza[buf][sr * LDT + sc]);
    async_b128(z + (size_t)(rowBase + sr + 64) * S_DIM + k0 + sc,
               &lza[buf][(sr + 64) * LDT + sc]);
    // W (Wout, bf16): 64 rows x 32 cols -> 256 chunks, 1 per thread
    async_b128(wob + (size_t)(jBase + sr) * S_DIM + k0 + sc,
               &lwo[buf][sr * LDT + sc]);
  };

  auto compute = [&](int buf) {
    Frag a[2], b[2];
#pragma unroll
    for (int ni = 0; ni < 2; ++ni) {
      const int wb = (wN * 32 + ni * 16 + bn) * LDT + bk16;
      b[ni].q[0] = *(const uint4*)&lwo[buf][wb];
      b[ni].q[1] = *(const uint4*)&lwo[buf][wb + 16];
    }
#pragma unroll
    for (int mi = 0; mi < 2; ++mi) {
      const int ab = (wM * 32 + mi * 16 + am) * LDT + ak;
      a[mi].q[0] = *(const uint4*)&lza[buf][ab];
      a[mi].q[1] = *(const uint4*)&lza[buf][ab + 16];
#pragma unroll
      for (int ni = 0; ni < 2; ++ni)
        acc[mi][ni] = wmma_bf16(a[mi].v, b[ni].v, acc[mi][ni]);
    }
  };

  stage(0, 0);
  for (int kt = 0; kt < NT; ++kt) {
    const int buf = kt & 1;
    wait_async0();
    __syncthreads();
    if (kt + 1 < NT) stage(kt + 1, buf ^ 1);
    compute(buf);
  }

  const int mh = (lane >> 4) * 8;
#pragma unroll
  for (int ni = 0; ni < 2; ++ni) {
    const int   j  = jBase + wN * 32 + ni * 16 + bn;
    const float bo = bout[j];
#pragma unroll
    for (int mi = 0; mi < 2; ++mi) {
#pragma unroll
      for (int vi = 0; vi < 8; ++vi) {
        const int row = rowBase + wM * 32 + mi * 16 + mh + vi;
        y[(size_t)row * S_DIM + j] = acc[mi][ni][vi] + bo;
      }
    }
  }
}

// ---------------------------------------------------------------------------
extern "C" void kernel_launch(void* const* d_in, const int* in_sizes, int n_in,
                              void* d_out, int out_size, void* d_ws, size_t ws_size,
                              hipStream_t stream) {
  (void)n_in; (void)out_size; (void)ws_size;
  const float* x        = (const float*)d_in[0];
  const float* last_x   = (const float*)d_in[1];
  const float* last_num = (const float*)d_in[2];
  const float* last_den = (const float*)d_in[3];
  const float* bonus    = (const float*)d_in[4];
  const float* mix_k    = (const float*)d_in[5];
  const float* mix_v    = (const float*)d_in[6];
  const float* mix_r    = (const float*)d_in[7];
  const float* Wk       = (const float*)d_in[8];
  const float* bk       = (const float*)d_in[9];
  const float* Wv       = (const float*)d_in[10];
  const float* bv       = (const float*)d_in[11];
  const float* Wr       = (const float*)d_in[12];
  const float* br       = (const float*)d_in[13];
  const float* Wout     = (const float*)d_in[14];
  const float* bout     = (const float*)d_in[15];

  const int rows = in_sizes[0] / S_DIM;            // 16384 tokens

  // workspace layout (bf16): z[rows*S] | Wk | Wv | Wr | Wout  (32MB + 4*2MB)
  unsigned short* z   = (unsigned short*)d_ws;
  unsigned short* wkb = z   + (size_t)rows * S_DIM;
  unsigned short* wvb = wkb + (size_t)S_DIM * S_DIM;
  unsigned short* wrb = wvb + (size_t)S_DIM * S_DIM;
  unsigned short* wob = wrb + (size_t)S_DIM * S_DIM;

  dim3 blk(256, 1, 1);

  const int cvtBlocks = (S_DIM * S_DIM) / (256 * 4);  // 1024
  cvt_bf16_kernel<<<cvtBlocks, blk, 0, stream>>>(Wk,   wkb);
  cvt_bf16_kernel<<<cvtBlocks, blk, 0, stream>>>(Wv,   wvb);
  cvt_bf16_kernel<<<cvtBlocks, blk, 0, stream>>>(Wr,   wrb);
  cvt_bf16_kernel<<<cvtBlocks, blk, 0, stream>>>(Wout, wob);

  dim3 g1(S_DIM / 32, rows / BM, 1);               // 32 x 128 blocks
  rwkv_kvr_kernel<<<g1, blk, 0, stream>>>(x, last_x, last_num, last_den, bonus,
                                          mix_k, mix_v, mix_r,
                                          wkb, wvb, wrb, bk, bv, br, z);

  dim3 g2(S_DIM / 64, rows / BM, 1);               // 16 x 128 blocks
  rwkv_out_kernel<<<g2, blk, 0, stream>>>(z, wob, bout, (float*)d_out);
}
